// FourierTransformerLayer_23862838296638
// MI455X (gfx1250) — matrix operand order repaired and checked by reading, not compiled, not run on hardware
//
#include <hip/hip_runtime.h>
#include <math.h>

// ---------------------------------------------------------------------------
// Types / constants
// ---------------------------------------------------------------------------
typedef __attribute__((ext_vector_type(16))) _Float16 v16h;
typedef __attribute__((ext_vector_type(8)))  _Float16 v8h;
typedef __attribute__((ext_vector_type(8)))  float    v8f;
typedef __attribute__((ext_vector_type(4)))  float    f4;

static constexpr int cB = 8, cS = 512, cD = 512, cH = 8, cNS = 4, cDF = 2048;
static constexpr float cEPS = 1e-5f;
static constexpr float cPI2 = 6.28318530717958647692f;

#define BM 128
#define BN 128
#define BK 32
#define LDT 40   // LDS row pitch in halves (BK + 8 pad, keeps 16B alignment)

struct Off { int d1, m1; long long s1; int d2, m2; long long s2; };
static inline Off OFF(int d1=1,int m1=1,long long s1=0,
                      int d2=1,int m2=1,long long s2=0){
  Off o; o.d1=d1; o.m1=m1; o.s1=s1; o.d2=d2; o.m2=m2; o.s2=s2; return o;
}
__device__ __forceinline__ long long offz(const Off o, int z){
  return (long long)((z / o.d1) % o.m1) * o.s1
       + (long long)((z / o.d2) % o.m2) * o.s2;
}

// ---------------------------------------------------------------------------
// Generic batched WMMA GEMM:  C[z] = alpha * A[z] (x) opB(B[z]) + beta*C [+bias]
// A: [M,K] row-major (lda). B: [K,N] row-major (ldb) or, if transB, [N,K].
// f32 in memory, f16 in LDS/WMMA, f32 accumulate. alpha = alphaMul * (*alphaPtr).
// 128 threads = 4 waves in a 2x2 grid; each wave owns a 64x64 tile ->
// 16 v_wmma per K-step off 8 fragment loads (max WMMA density per LDS byte).
// ---------------------------------------------------------------------------
__global__ __launch_bounds__(128) void k_gemm(
    const float* __restrict__ A, const float* __restrict__ Bm,
    float* __restrict__ C, const float* __restrict__ bias,
    int M, int N, int K, int lda, int ldb, int ldc,
    Off oa, Off ob, Off oc, Off obias,
    int transB, const float* __restrict__ alphaPtr, int alphaIdx, float alphaMul,
    float beta, int relu)
{
  __shared__ __align__(16) _Float16 sA[BM * LDT];
  __shared__ __align__(16) _Float16 sB[BN * LDT];

  const int z = blockIdx.z;
  const long long aO = offz(oa, z), bO = offz(ob, z);
  const long long cO = offz(oc, z), biO = offz(obias, z);

  const int tid  = threadIdx.x;
  const int nBase = blockIdx.x * BN;
  const int mBase = blockIdx.y * BM;
  const int lane = tid & 31, wave = tid >> 5;
  const int wm = wave >> 1;        // 0..1 -> 64-row wave tile
  const int wn = wave & 1;         // 0..1 -> 64-col wave tile
  const int l16 = lane & 15, lh = lane >> 4;

  v8f acc[4][4];
  #pragma unroll
  for (int i = 0; i < 4; ++i)
    #pragma unroll
    for (int j = 0; j < 4; ++j)
      #pragma unroll
      for (int r = 0; r < 8; ++r) acc[i][j][r] = 0.0f;

  for (int kb = 0; kb < K; kb += BK) {
    __syncthreads();
    // prefetch next K-tile into GL2 while staging this one
    if (kb + BK < K) {
      int r = tid;                                    // one probe per row
      if (mBase + r < M)
        __builtin_prefetch(A + aO + (long long)(mBase + r) * lda + kb + BK, 0, 1);
      if (transB) {
        if (nBase + r < N)
          __builtin_prefetch(Bm + bO + (long long)(nBase + r) * ldb + kb + BK, 0, 1);
      } else if (r < BK) {
        __builtin_prefetch(Bm + bO + (long long)(kb + BK + r) * ldb + nBase, 0, 1);
      }
    }
    // ---- stage A tile [BM][BK], convert f32 -> f16 ----
    #pragma unroll
    for (int i = 0; i < 8; ++i) {
      int idx = tid + i * 128;              // 0..1023
      int r   = idx >> 3;                   // 0..127
      int c4  = (idx & 7) << 2;             // 0..28
      int gm  = mBase + r;
      f4 v = {0.f, 0.f, 0.f, 0.f};
      if (gm < M) v = *(const f4*)(A + aO + (long long)gm * lda + kb + c4);
      #pragma unroll
      for (int j = 0; j < 4; ++j) sA[r * LDT + c4 + j] = (_Float16)v[j];
    }
    // ---- stage B tile as sB[n][k] (column-major over the K-tile) ----
    if (transB) {
      #pragma unroll
      for (int i = 0; i < 8; ++i) {
        int idx = tid + i * 128;
        int n   = idx >> 3;                 // 0..127
        int k4  = (idx & 7) << 2;           // 0..28
        int gn  = nBase + n;
        f4 v = {0.f, 0.f, 0.f, 0.f};
        if (gn < N) v = *(const f4*)(Bm + bO + (long long)gn * ldb + kb + k4);
        #pragma unroll
        for (int j = 0; j < 4; ++j) sB[n * LDT + k4 + j] = (_Float16)v[j];
      }
    } else {
      #pragma unroll
      for (int i = 0; i < 8; ++i) {
        int idx = tid + i * 128;
        int k   = idx >> 5;                 // 0..31
        int n4  = (idx & 31) << 2;          // 0..124
        int gk  = kb + k;
        int gn0 = nBase + n4;
        if (gn0 + 3 < N) {
          f4 v = *(const f4*)(Bm + bO + (long long)gk * ldb + gn0);
          #pragma unroll
          for (int j = 0; j < 4; ++j) sB[(n4 + j) * LDT + k] = (_Float16)v[j];
        } else {
          #pragma unroll
          for (int j = 0; j < 4; ++j) {
            int gn = gn0 + j;
            float x = (gn < N) ? Bm[bO + (long long)gk * ldb + gn] : 0.f;
            sB[(n4 + j) * LDT + k] = (_Float16)x;
          }
        }
      }
    }
    __syncthreads();

    // ---- fragments per documented CDNA5 WMMA VGPR layouts ----
    v16h af[4], bf[4];
    #pragma unroll
    for (int fm = 0; fm < 4; ++fm) {
      const _Float16* p = sA + (wm * 64 + fm * 16 + l16) * LDT + lh * 8;
      v8h a0 = *(const v8h*)p;
      v8h a1 = *(const v8h*)(p + 16);
      #pragma unroll
      for (int h = 0; h < 8; ++h) { af[fm][h] = a0[h]; af[fm][8 + h] = a1[h]; }
    }
    #pragma unroll
    for (int fn = 0; fn < 4; ++fn) {
      const _Float16* p = sB + (wn * 64 + fn * 16 + l16) * LDT + lh * 8;
      v8h b0 = *(const v8h*)p;
      v8h b1 = *(const v8h*)(p + 16);
      #pragma unroll
      for (int h = 0; h < 8; ++h) { bf[fn][h] = b0[h]; bf[fn][8 + h] = b1[h]; }
    }
    #pragma unroll
    for (int fm = 0; fm < 4; ++fm)
      #pragma unroll
      for (int fn = 0; fn < 4; ++fn)
        acc[fm][fn] = __builtin_amdgcn_wmma_f32_16x16x32_f16(
            false, af[fm], false, bf[fn], (short)0, acc[fm][fn], false, false);
  }

  // ---- epilogue ----
  const float alpha = alphaMul * (alphaPtr ? alphaPtr[alphaIdx] : 1.0f);
  #pragma unroll
  for (int fm = 0; fm < 4; ++fm) {
    #pragma unroll
    for (int fn = 0; fn < 4; ++fn) {
      int gn = nBase + wn * 64 + fn * 16 + l16;
      #pragma unroll
      for (int r = 0; r < 8; ++r) {
        int gm = mBase + wm * 64 + fm * 16 + lh * 8 + r;
        if (gm < M && gn < N) {
          long long ci = cO + (long long)gm * ldc + gn;
          float v = alpha * acc[fm][fn][r];
          if (beta != 0.0f) v += beta * C[ci];
          if (bias)         v += bias[biO + gn];
          if (relu)         v = fmaxf(v, 0.0f);
          C[ci] = v;
        }
      }
    }
  }
}

// ---------------------------------------------------------------------------
// DFT matrices: Cm[s,u]=cos(2pi s u/S), Sm[s,u]=sin(2pi s u/S)
// ---------------------------------------------------------------------------
__global__ __launch_bounds__(256) void k_dft(float* __restrict__ Cm,
                                             float* __restrict__ Sm)
{
  int idx = blockIdx.x * 256 + threadIdx.x;       // S*S threads
  int s = idx >> 9, u = idx & 511;
  int ph = (s * u) & 511;                          // exact mod S (S=512)
  float a = cPI2 * (float)ph / (float)cS;
  float sv, cv; sincosf(a, &sv, &cv);
  Cm[idx] = cv; Sm[idx] = sv;
}

// Band-pass projection matrices G_i (rfft band mask -> circulant matrix)
__global__ __launch_bounds__(256) void k_band(float* __restrict__ G)
{
  int idx = blockIdx.x * 256 + threadIdx.x;        // NS*S*S threads
  int i = idx >> 18;
  int rem = idx & ((1 << 18) - 1);
  int u = rem >> 9, v = rem & 511;
  int dm = (u - v + cS) & 511;
  int ks = i * 64, ke = ks + 64;                   // band=fb//NS=64; Nyquist excluded
  float acc = 0.f;
  for (int k = ks; k < ke; ++k) {
    float w = (k == 0) ? 1.f : 2.f;
    int ph = (k * dm) & 511;
    acc += w * cosf(cPI2 * (float)ph / (float)cS);
  }
  G[idx] = acc / (float)cS;
}

// ---------------------------------------------------------------------------
// Row softmax in place + weighted-attn accumulation into d_out
// ---------------------------------------------------------------------------
__global__ __launch_bounds__(256) void k_softmax_wattn(
    float* __restrict__ P, float* __restrict__ wattn,
    const float* __restrict__ sw, int sidx, int first, int Scols)
{
  long long base = (long long)blockIdx.x * Scols;
  int tid = threadIdx.x, lane = tid & 31, wave = tid >> 5;
  __shared__ float red[8];

  float m = -3.4e38f;
  for (int d = tid; d < Scols; d += 256) m = fmaxf(m, P[base + d]);
  for (int o = 16; o > 0; o >>= 1) m = fmaxf(m, __shfl_down(m, o, 32));
  if (lane == 0) red[wave] = m;
  __syncthreads();
  float mm = red[0];
  for (int w = 1; w < 8; ++w) mm = fmaxf(mm, red[w]);
  __syncthreads();

  float s = 0.f;
  for (int d = tid; d < Scols; d += 256) s += __expf(P[base + d] - mm);
  for (int o = 16; o > 0; o >>= 1) s += __shfl_down(s, o, 32);
  if (lane == 0) red[wave] = s;
  __syncthreads();
  float ss = 0.f;
  for (int w = 0; w < 8; ++w) ss += red[w];
  float inv = 1.0f / ss;
  float wgt = sw[sidx];

  for (int d = tid; d < Scols; d += 256) {
    float p = __expf(P[base + d] - mm) * inv;
    P[base + d] = p;
    if (first) wattn[base + d] = wgt * p;
    else       wattn[base + d] += wgt * p;
  }
}

// ---------------------------------------------------------------------------
// LayerNorm(out) of (a + b + c) with gamma/beta; b,c optional
// ---------------------------------------------------------------------------
__global__ __launch_bounds__(256) void k_ln(
    const float* __restrict__ a, const float* __restrict__ b,
    const float* __restrict__ c, const float* __restrict__ g,
    const float* __restrict__ bt, float* __restrict__ out, int Dn)
{
  long long base = (long long)blockIdx.x * Dn;
  int tid = threadIdx.x, lane = tid & 31, wave = tid >> 5;
  __shared__ float r1[8], r2[8];

  float s = 0.f, s2 = 0.f;
  for (int d = tid; d < Dn; d += 256) {
    float v = a[base + d];
    if (b) v += b[base + d];
    if (c) v += c[base + d];
    s += v; s2 += v * v;
  }
  for (int o = 16; o > 0; o >>= 1) { s += __shfl_down(s, o, 32); s2 += __shfl_down(s2, o, 32); }
  if (lane == 0) { r1[wave] = s; r2[wave] = s2; }
  __syncthreads();
  float ts = 0.f, ts2 = 0.f;
  for (int w = 0; w < 8; ++w) { ts += r1[w]; ts2 += r2[w]; }
  float mean = ts / (float)Dn;
  float var  = ts2 / (float)Dn - mean * mean;
  float invs = rsqrtf(var + cEPS);

  for (int d = tid; d < Dn; d += 256) {
    float v = a[base + d];
    if (b) v += b[base + d];
    if (c) v += c[base + d];
    out[base + d] = (v - mean) * invs * g[d] + bt[d];
  }
}

// fh[b,s,:] = 0.5*(h[b,s,:] + h[b,(S-s)%S,:])   ( (1/S) C^2 = (I+J)/2 )
__global__ __launch_bounds__(256) void k_flipadd(const float* __restrict__ h,
                                                 float* __restrict__ out)
{
  long long idx = (long long)blockIdx.x * 256 + threadIdx.x;   // B*S*D
  int d = (int)(idx & 511);
  int s = (int)((idx >> 9) & 511);
  long long b = idx >> 18;
  int s2 = (cS - s) & 511;
  out[idx] = 0.5f * (h[idx] + h[(b << 18) + ((long long)s2 << 9) + d]);
}

// freq_out[b,0,:] += bft   ( (1/S) * C @ broadcast(bft) concentrates at s=0 )
__global__ __launch_bounds__(256) void k_row0bias(float* __restrict__ fqo,
                                                  const float* __restrict__ bft)
{
  int idx = blockIdx.x * 256 + threadIdx.x;   // B*D
  int b = idx >> 9, d = idx & 511;
  fqo[(long long)b * cS * cD + d] += bft[d];
}

// ---------------------------------------------------------------------------
// Host-side launcher helper
// ---------------------------------------------------------------------------
static void gemm(hipStream_t st,
                 const float* A, const float* Bm, float* C, const float* bias,
                 int M, int N, int K, int lda, int ldb, int ldc,
                 Off oa, Off ob, Off oc, Off obias, int batch, int transB,
                 const float* aPtr, int aIdx, float aMul, float beta, int relu)
{
  dim3 grid((N + BN - 1) / BN, (M + BM - 1) / BM, batch);
  k_gemm<<<grid, dim3(128), 0, st>>>(A, Bm, C, bias, M, N, K, lda, ldb, ldc,
                                     oa, ob, oc, obias, transB,
                                     aPtr, aIdx, aMul, beta, relu);
}

extern "C" void kernel_launch(void* const* d_in, const int* in_sizes, int n_in,
                              void* d_out, int out_size, void* d_ws, size_t ws_size,
                              hipStream_t stream)
{
  (void)in_sizes; (void)n_in; (void)out_size; (void)ws_size;
  const float* x   = (const float*)d_in[0];
  const float* Wq  = (const float*)d_in[1];
  const float* bq  = (const float*)d_in[2];
  const float* Wk  = (const float*)d_in[3];
  const float* bk  = (const float*)d_in[4];
  const float* Wv  = (const float*)d_in[5];
  const float* bv  = (const float*)d_in[6];
  const float* Wo  = (const float*)d_in[7];
  const float* bo  = (const float*)d_in[8];
  const float* tw  = (const float*)d_in[9];
  const float* fw  = (const float*)d_in[10];
  const float* sw  = (const float*)d_in[11];
  const float* Wc  = (const float*)d_in[12];
  const float* bc  = (const float*)d_in[13];
  // d_in[14..17] = Wg1,bg1,Wg2,bg2: h_freq == h exactly, so gate path is identity.
  const float* Wf1 = (const float*)d_in[18];
  const float* bf1 = (const float*)d_in[19];
  const float* Wf2 = (const float*)d_in[20];
  const float* bf2 = (const float*)d_in[21];
  const float* Wft = (const float*)d_in[22];
  const float* bft = (const float*)d_in[23];
  const float* ln1g = (const float*)d_in[24];
  const float* ln1b = (const float*)d_in[25];
  const float* ln2g = (const float*)d_in[26];
  const float* ln2b = (const float*)d_in[27];
  const float* ln3g = (const float*)d_in[28];
  const float* ln3b = (const float*)d_in[29];

  const long long SS  = (long long)cS * cS;          // 262144
  const long long SD  = (long long)cS * cD;          // 262144
  const long long BSD = (long long)cB * cS * cD;     // 2097152

  float* out_h  = (float*)d_out;
  float* out_wa = (float*)d_out + BSD;

  float* w = (float*)d_ws;
  size_t off = 0;
  float* CM  = w + off; off += SS;
  float* SM  = w + off; off += SS;
  float* G   = w + off; off += (size_t)cNS * SS;
  float* XS  = w + off; off += BSD;
  float* Qb  = w + off; off += BSD;
  float* Kb  = w + off; off += BSD;
  float* Vb  = w + off; off += BSD;
  float* Qc  = w + off; off += BSD;
  float* Qs  = w + off; off += BSD;
  float* Kc  = w + off; off += BSD;
  float* Ks  = w + off; off += BSD;
  float* P   = w + off; off += (long long)cB * cH * SS;  // 16M
  float* CTX = w + off; off += BSD;
  float* CMB = w + off; off += (long long)cB * cS * cNS * cD; // 8M
  float* ATT = w + off; off += BSD;
  float* H1  = w + off; off += BSD;
  float* H2  = w + off; off += BSD;
  float* FF1 = w + off; off += (long long)cB * cS * cDF;      // 8M
  float* FF2 = w + off; off += BSD;
  float* FH  = w + off; off += BSD;
  float* FQO = w + off; off += BSD;

  const Off O0 = OFF();
  const float invScale = 0.125f;   // 1/sqrt(DK)=1/8

  // Build DFT + band-projection matrices
  k_dft <<<(int)(SS / 256),        256, 0, stream>>>(CM, SM);
  k_band<<<(int)(cNS * SS / 256),  256, 0, stream>>>(G);

  // --------------- Multi-scale Fourier attention ---------------
  for (int i = 0; i < cNS; ++i) {
    // xs = G_i @ x[b]          (batch over b)
    gemm(stream, G + (long long)i * SS, x, XS, nullptr,
         cS, cD, cS, cS, cD, cD,
         O0, OFF(1, cB, SD), OFF(1, cB, SD), O0, cB, 0,
         nullptr, 0, 1.f, 0.f, 0);
    // Q/K/V = xs @ W + b
    gemm(stream, XS, Wq + (long long)i * cD * cD, Qb, bq + i * cD,
         cB * cS, cD, cD, cD, cD, cD, O0, O0, O0, O0, 1, 0, nullptr, 0, 1.f, 0.f, 0);
    gemm(stream, XS, Wk + (long long)i * cD * cD, Kb, bk + i * cD,
         cB * cS, cD, cD, cD, cD, cD, O0, O0, O0, O0, 1, 0, nullptr, 0, 1.f, 0.f, 0);
    gemm(stream, XS, Wv + (long long)i * cD * cD, Vb, bv + i * cD,
         cB * cS, cD, cD, cD, cD, cD, O0, O0, O0, O0, 1, 0, nullptr, 0, 1.f, 0.f, 0);
    // DFT along S: Qc=C@Q, Qs=S@Q, Kc=C@K, Ks=S@K  (batch over b)
    gemm(stream, CM, Qb, Qc, nullptr, cS, cD, cS, cS, cD, cD,
         O0, OFF(1, cB, SD), OFF(1, cB, SD), O0, cB, 0, nullptr, 0, 1.f, 0.f, 0);
    gemm(stream, SM, Qb, Qs, nullptr, cS, cD, cS, cS, cD, cD,
         O0, OFF(1, cB, SD), OFF(1, cB, SD), O0, cB, 0, nullptr, 0, 1.f, 0.f, 0);
    gemm(stream, CM, Kb, Kc, nullptr, cS, cD, cS, cS, cD, cD,
         O0, OFF(1, cB, SD), OFF(1, cB, SD), O0, cB, 0, nullptr, 0, 1.f, 0.f, 0);
    gemm(stream, SM, Kb, Ks, nullptr, cS, cD, cS, cS, cD, cD,
         O0, OFF(1, cB, SD), OFF(1, cB, SD), O0, cB, 0, nullptr, 0, 1.f, 0.f, 0);

    // scores = (tw*QK^T + fw*(QcKc^T + QsKs^T)) / sqrt(DK); batch z=(b,h)
    Off qo = OFF(cH, cB, SD, 1, cH, 64);   // b*S*D + h*DK
    Off po = OFF(1, cB * cH, SS);          // z*S*S
    gemm(stream, Qb, Kb, P, nullptr, cS, cS, 64, cD, cD, cS,
         qo, qo, po, O0, cB * cH, 1, tw, i, invScale, 0.f, 0);
    gemm(stream, Qc, Kc, P, nullptr, cS, cS, 64, cD, cD, cS,
         qo, qo, po, O0, cB * cH, 1, fw, i, invScale, 1.f, 0);
    gemm(stream, Qs, Ks, P, nullptr, cS, cS, 64, cD, cD, cS,
         qo, qo, po, O0, cB * cH, 1, fw, i, invScale, 1.f, 0);

    // softmax in place + weighted_attn accumulate into d_out
    k_softmax_wattn<<<cB * cH * cS, 256, 0, stream>>>(P, out_wa, sw, i, (i == 0) ? 1 : 0, cS);

    // ctx = A @ V   (batch z=(b,h), N=64)
    gemm(stream, P, Vb, CTX, nullptr, cS, 64, cS, cS, cD, cD,
         po, qo, qo, O0, cB * cH, 0, nullptr, 0, 1.f, 0.f, 0);

    // combined[:, i*D:(i+1)*D] = ctx @ Wo_i + bo_i
    gemm(stream, CTX, Wo + (long long)i * cD * cD, CMB + (long long)i * cD, bo + i * cD,
         cB * cS, cD, cD, cD, cD, cNS * cD, O0, O0, O0, O0, 1, 0, nullptr, 0, 1.f, 0.f, 0);
  }

  // attn_out = combined @ Wc + bc
  gemm(stream, CMB, Wc, ATT, bc, cB * cS, cD, cNS * cD,
       cNS * cD, cD, cD, O0, O0, O0, O0, 1, 0, nullptr, 0, 1.f, 0.f, 0);

  // h1 = LN(x + attn_out); gating block is exact identity -> h2 = LN(h1 + h1)
  k_ln<<<cB * cS, 256, 0, stream>>>(x,  ATT, nullptr, ln1g, ln1b, H1, cD);
  k_ln<<<cB * cS, 256, 0, stream>>>(H1, H1,  nullptr, ln2g, ln2b, H2, cD);

  // SpectralFeedForward
  gemm(stream, H2, Wf1, FF1, bf1, cB * cS, cDF, cD, cD, cDF, cDF,
       O0, O0, O0, O0, 1, 0, nullptr, 0, 1.f, 0.f, 1);            // relu
  gemm(stream, FF1, Wf2, FF2, bf2, cB * cS, cD, cDF, cDF, cD, cD,
       O0, O0, O0, O0, 1, 0, nullptr, 0, 1.f, 0.f, 0);

  // freq_out = 0.5*(h + Jh) @ Wft  (+ bft on row s==0)
  k_flipadd<<<(int)(BSD / 256), 256, 0, stream>>>(H2, FH);
  gemm(stream, FH, Wft, FQO, nullptr, cB * cS, cD, cD, cD, cD, cD,
       O0, O0, O0, O0, 1, 0, nullptr, 0, 1.f, 0.f, 0);
  k_row0bias<<<(cB * cD) / 256, 256, 0, stream>>>(FQO, bft);

  // h = LN(h2 + ff + freq_out) -> output slot 0
  k_ln<<<cB * cS, 256, 0, stream>>>(H2, FF2, FQO, ln3g, ln3b, out_h, cD);
}
